// AttnAdapter_49709951484644
// MI455X (gfx1250) — matrix operand order repaired and checked by reading, not compiled, not run on hardware
//
#include <hip/hip_runtime.h>
#include <math.h>

// Problem constants
#define S_LEN   2048
#define D_MODEL 4096
#define NH      32
#define NKV     8
#define HEADD   128
#define NQKV    6144   // H*HD + 2*KV*HD
#define BOUND   611    // SYS_LEN + IMG_LEN
#define SYSL    35

#define USE_ASYNC_LDS 1

typedef __attribute__((ext_vector_type(16))) __bf16 v16bf;
typedef __attribute__((ext_vector_type(8)))  __bf16 v8bf;
typedef __attribute__((ext_vector_type(8)))  float  v8f;
typedef __attribute__((address_space(3)))    __bf16 lds_bf16;

static __device__ __forceinline__ unsigned short f2bfu(float f) {
    unsigned u = __float_as_uint(f);
    u += 0x7FFFu + ((u >> 16) & 1u);   // round-to-nearest-even
    return (unsigned short)(u >> 16);
}
static __device__ __forceinline__ __bf16 f2bf(float f) {
    unsigned short h = f2bfu(f);
    return __builtin_bit_cast(__bf16, h);
}
static __device__ __forceinline__ v16bf cat8(v8bf lo, v8bf hi) {
    v16bf r;
#pragma unroll
    for (int i = 0; i < 8; ++i) { r[i] = lo[i]; r[i + 8] = hi[i]; }
    return r;
}
static __device__ __forceinline__ v8bf ld8(const __bf16* p) {
    return *(const v8bf*)p;
}
static __device__ __forceinline__ v8f wmma_bf16(v16bf a, v16bf b, v8f c) {
    return __builtin_amdgcn_wmma_f32_16x16x32_bf16(false, a, false, b, (short)0, c,
                                                   false, false);
}

#if USE_ASYNC_LDS
// Async copy 16B global -> LDS (gfx1250 GLOBAL_LOAD_ASYNC_TO_LDS_B128, ASYNCcnt).
static __device__ __forceinline__ void async_b128(const __bf16* g, __bf16* l) {
    unsigned loff = (unsigned)(uintptr_t)(lds_bf16*)l;  // LDS byte offset
    asm volatile("global_load_async_to_lds_b128 %0, %1, off"
                 :: "v"(loff), "v"(g) : "memory");
}
static __device__ __forceinline__ void wait_async8() {
    asm volatile("s_wait_asynccnt 0x8" ::: "memory");
}
static __device__ __forceinline__ void wait_async0() {
    asm volatile("s_wait_asynccnt 0x0" ::: "memory");
}
#endif

// Stage one thread's share of a 128x64 A tile + 128x64 B tile into LDS.
// Each thread covers 64B (32 elems) of one A row and one B row.
static __device__ __forceinline__ void stage_tile(const __bf16* ga, const __bf16* gb,
                                                  __bf16* la, __bf16* lb) {
#if USE_ASYNC_LDS
    async_b128(ga +  0, la +  0); async_b128(ga +  8, la +  8);
    async_b128(ga + 16, la + 16); async_b128(ga + 24, la + 24);
    async_b128(gb +  0, lb +  0); async_b128(gb +  8, lb +  8);
    async_b128(gb + 16, lb + 16); async_b128(gb + 24, lb + 24);
#else
    *(v8bf*)(la +  0) = ld8(ga +  0); *(v8bf*)(la +  8) = ld8(ga +  8);
    *(v8bf*)(la + 16) = ld8(ga + 16); *(v8bf*)(la + 24) = ld8(ga + 24);
    *(v8bf*)(lb +  0) = ld8(gb +  0); *(v8bf*)(lb +  8) = ld8(gb +  8);
    *(v8bf*)(lb + 16) = ld8(gb + 16); *(v8bf*)(lb + 24) = ld8(gb + 24);
#endif
}

// ---------------------------------------------------------------- convert
__global__ void cvt_f32_bf16(const float* __restrict__ src,
                             __bf16* __restrict__ dst, long long n) {
    long long i = ((long long)blockIdx.x * blockDim.x + threadIdx.x) * 4;
    if (i >= n) return;
    float4 f = *(const float4*)(src + i);
    uint2 o;
    o.x = (unsigned)f2bfu(f.x) | ((unsigned)f2bfu(f.y) << 16);
    o.y = (unsigned)f2bfu(f.z) | ((unsigned)f2bfu(f.w) << 16);
    *(uint2*)((char*)dst + i * 2) = o;
}

// ---------------------------------------------------------- NT bf16 GEMM
// C[M,N] = A[M,K] * B[N,K]^T  (row-major, K contiguous).
// 256 threads (8 waves); block tile 128x128; K-step 64; LDS double-buffered
// via async global->LDS; wave tile 32x64 (2x4 WMMA accumulators).
#define LDS_STRIDE 72   // 64 elems + 8 pad (bank spread), 16B-aligned rows
__global__ void __launch_bounds__(256, 1)
gemm_bf16nt(const __bf16* __restrict__ A, const __bf16* __restrict__ Bm,
            float* __restrict__ C, int M, int N, int K) {
    __shared__ __align__(16) __bf16 Al[2][128 * LDS_STRIDE];
    __shared__ __align__(16) __bf16 Bl[2][128 * LDS_STRIDE];

    const int tid = threadIdx.x;
    const int lane = tid & 31, wave = tid >> 5;
    const int wm = wave & 3, wn = wave >> 2;     // 4 waves M x 2 waves N
    const int l15 = lane & 15, sel = lane >> 4;
    const int mBlk = blockIdx.y * 128, nBlk = blockIdx.x * 128;

    // staging coords: 2 threads per row, 32 elems each
    const int sRow = tid >> 1;
    const int sCol = (tid & 1) * 32;
    const __bf16* gA0 = A + (size_t)(mBlk + sRow) * K + sCol;
    const __bf16* gB0 = Bm + (size_t)(nBlk + sRow) * K + sCol;
    __bf16* lA0 = &Al[0][sRow * LDS_STRIDE + sCol];
    __bf16* lB0 = &Bl[0][sRow * LDS_STRIDE + sCol];
    __bf16* lA1 = &Al[1][sRow * LDS_STRIDE + sCol];
    __bf16* lB1 = &Bl[1][sRow * LDS_STRIDE + sCol];

    v8f acc[2][4];
#pragma unroll
    for (int i = 0; i < 2; ++i)
#pragma unroll
        for (int j = 0; j < 4; ++j) acc[i][j] = v8f{};

    const int nStage = K / 64;
    stage_tile(gA0, gB0, lA0, lB0);

    for (int ks = 0; ks < nStage; ++ks) {
        const int buf = ks & 1;
        if (ks + 1 < nStage)
            stage_tile(gA0 + (ks + 1) * 64, gB0 + (ks + 1) * 64,
                       buf ? lA0 : lA1, buf ? lB0 : lB1);
#if USE_ASYNC_LDS
        if (ks + 1 < nStage) wait_async8(); else wait_async0();
#endif
        __syncthreads();   // tile[buf] fully resident for all waves

        const __bf16* Ab_ = &Al[buf][0];
        const __bf16* Bb_ = &Bl[buf][0];
#pragma unroll
        for (int ksub = 0; ksub < 64; ksub += 32) {
            const int a0o = (wm * 32 + l15) * LDS_STRIDE + ksub + sel * 8;
            v16bf a0 = cat8(ld8(Ab_ + a0o), ld8(Ab_ + a0o + 16));
            v16bf a1 = cat8(ld8(Ab_ + a0o + 16 * LDS_STRIDE),
                            ld8(Ab_ + a0o + 16 * LDS_STRIDE + 16));
#pragma unroll
            for (int j = 0; j < 4; ++j) {
                const int bo = (wn * 64 + j * 16 + l15) * LDS_STRIDE + ksub + sel * 8;
                v16bf b = cat8(ld8(Bb_ + bo), ld8(Bb_ + bo + 16));
                acc[0][j] = wmma_bf16(a0, b, acc[0][j]);
                acc[1][j] = wmma_bf16(a1, b, acc[1][j]);
            }
        }
        __syncthreads();   // all waves done reading before buffer is overwritten
    }

#pragma unroll
    for (int i = 0; i < 2; ++i)
#pragma unroll
        for (int j = 0; j < 4; ++j) {
            int n = nBlk + wn * 64 + j * 16 + l15;
#pragma unroll
            for (int r = 0; r < 8; ++r) {
                int m = mBlk + wm * 32 + i * 16 + r + sel * 8;
                C[(size_t)m * N + n] = acc[i][j][r];
            }
        }
}

// ------------------------------------------------------- RoPE + reorder
// qkv fp32 [S][6144]  ->  Qb[H][S][128], Kb[KV][S][128], Vt[KV][128][S] (bf16)
__global__ void rope_reorder(const float* __restrict__ qkv,
                             __bf16* __restrict__ Qb, __bf16* __restrict__ Kb,
                             __bf16* __restrict__ Vt) {
    long long idx = (long long)blockIdx.x * blockDim.x + threadIdx.x;
    const long long total = (long long)S_LEN * NQKV;
    if (idx >= total) return;
    int c = (int)(idx % NQKV);
    int s = (int)(idx / NQKV);
    const float* row = qkv + (size_t)s * NQKV;
    if (c < D_MODEL + 1024) {  // q or k: apply RoPE
        int base = (c < D_MODEL) ? 0 : D_MODEL;
        int cc = c - base;
        int head = cc / HEADD;
        int d = cc % HEADD;
        int j = (d < 64) ? d : d - 64;
        float inv = __expf(-(float)j * (9.210340371976184f / 64.0f)); // 10000^(-j/64)
        float ang = (float)s * inv;
        float sn, cs;
        __sincosf(ang, &sn, &cs);
        float x = row[c];
        float xr = (d < 64) ? -row[c + 64] : row[c - 64];
        float y = x * cs + xr * sn;
        if (base == 0)
            Qb[((size_t)head * S_LEN + s) * HEADD + d] = f2bf(y);
        else
            Kb[((size_t)head * S_LEN + s) * HEADD + d] = f2bf(y);
    } else {  // v: transpose to [KV][HD][S]
        int cc = c - (D_MODEL + 1024);
        int head = cc / HEADD;
        int d = cc % HEADD;
        Vt[((size_t)head * HEADD + d) * S_LEN + s] = f2bf(row[c]);
    }
}

// ----------------------------------------------------- flash attention
// one wave per (16-query tile, head); online softmax over 32-key blocks.
__global__ void __launch_bounds__(32, 1)
attn_fwd(const __bf16* __restrict__ Qb, const __bf16* __restrict__ Kb,
         const __bf16* __restrict__ Vt, __bf16* __restrict__ Ob) {
    __shared__ __align__(64) __bf16 Plds[16 * 32];
    const int qB = blockIdx.x * 16;
    const int h = blockIdx.y;
    const int kv = h >> 2;  // GROUPS = 4
    const int lane = threadIdx.x;
    const int l15 = lane & 15;
    const int sel = lane >> 4;

    const __bf16* Qrow = Qb + ((size_t)h * S_LEN + (qB + l15)) * HEADD;
    v16bf aq[4];
#pragma unroll
    for (int c = 0; c < 4; ++c) {
        int ko = c * 32 + sel * 8;
        aq[c] = cat8(ld8(Qrow + ko), ld8(Qrow + ko + 16));
    }

    float rmax[8], rsum[8];
    v8f o[8];
#pragma unroll
    for (int r = 0; r < 8; ++r) { rmax[r] = -3.0e38f; rsum[r] = 0.0f; }
#pragma unroll
    for (int t = 0; t < 8; ++t) o[t] = v8f{};

    const float sm_scale = 0.08838834764831845f;  // 1/sqrt(128)

    for (int kb = 0; kb < qB + 16; kb += 32) {
        float p[2][8];
#pragma unroll
        for (int t = 0; t < 2; ++t) {
            int key = kb + t * 16 + l15;
            const __bf16* Krow = Kb + ((size_t)kv * S_LEN + key) * HEADD;
            v8f st = v8f{};
#pragma unroll
            for (int c = 0; c < 4; ++c) {
                int ko = c * 32 + sel * 8;
                v16bf bk = cat8(ld8(Krow + ko), ld8(Krow + ko + 16));
                st = wmma_bf16(aq[c], bk, st);
            }
            float ks = (key < SYSL) ? 0.5f : ((key < BOUND) ? 1.5f : 1.0f);
#pragma unroll
            for (int r = 0; r < 8; ++r) {
                int q = qB + r + sel * 8;
                float sc = st[r] * sm_scale;
                if (q >= BOUND) sc *= ks;
                p[t][r] = (key <= q) ? sc : -3.0e38f;
            }
        }
#pragma unroll
        for (int r = 0; r < 8; ++r) {
            float tm = fmaxf(p[0][r], p[1][r]);
#pragma unroll
            for (int off = 1; off < 16; off <<= 1)
                tm = fmaxf(tm, __shfl_xor(tm, off, 32));
            float nm = fmaxf(rmax[r], tm);
            float corr = __expf(rmax[r] - nm);
            rmax[r] = nm;
            float e0 = __expf(p[0][r] - nm);
            float e1 = __expf(p[1][r] - nm);
            p[0][r] = e0; p[1][r] = e1;
            float ts = e0 + e1;
#pragma unroll
            for (int off = 1; off < 16; off <<= 1)
                ts += __shfl_xor(ts, off, 32);
            rsum[r] = rsum[r] * corr + ts;
#pragma unroll
            for (int t = 0; t < 8; ++t) o[t][r] *= corr;
        }
        // transpose P: C-layout -> A-layout via LDS
        __syncthreads();
#pragma unroll
        for (int r = 0; r < 8; ++r) {
            int row = r + sel * 8;
            Plds[row * 32 + l15]      = f2bf(p[0][r]);
            Plds[row * 32 + 16 + l15] = f2bf(p[1][r]);
        }
        __syncthreads();
        int ko = sel * 8;
        v16bf pa = cat8(ld8(&Plds[l15 * 32 + ko]), ld8(&Plds[l15 * 32 + ko + 16]));
#pragma unroll
        for (int t = 0; t < 8; ++t) {
            const __bf16* Vrow =
                Vt + ((size_t)kv * HEADD + t * 16 + l15) * S_LEN + kb + sel * 8;
            v16bf bv = cat8(ld8(Vrow), ld8(Vrow + 16));
            o[t] = wmma_bf16(pa, bv, o[t]);
        }
    }
#pragma unroll
    for (int t = 0; t < 8; ++t)
#pragma unroll
        for (int r = 0; r < 8; ++r) {
            int m = qB + r + sel * 8;
            int d = t * 16 + l15;
            Ob[(size_t)m * D_MODEL + h * HEADD + d] = f2bf(o[t][r] / rsum[r]);
        }
}

// ---------------------------------------------------------------- launch
extern "C" void kernel_launch(void* const* d_in, const int* in_sizes, int n_in,
                              void* d_out, int out_size, void* d_ws, size_t ws_size,
                              hipStream_t stream) {
    const float* hidden = (const float*)d_in[0];
    const float* wq = (const float*)d_in[1];
    const float* wk = (const float*)d_in[2];
    const float* wv = (const float*)d_in[3];
    const float* wo = (const float*)d_in[4];

    char* ws = (char*)d_ws;
    __bf16* Xb    = (__bf16*)(ws);                       // 2048x4096 bf16 (16 MB)
    __bf16* Wqkvb = (__bf16*)(ws + (size_t)16777216);    // 6144x4096 bf16 (48 MB)
    __bf16* Wob   = (__bf16*)(ws + (size_t)67108864);    // 4096x4096 bf16 (32 MB)
    float*  QKV   = (float*) (ws + (size_t)100663296);   // 2048x6144 f32 (48 MB)
    __bf16* Qb    = (__bf16*)(ws + (size_t)150994944);   // 32x2048x128 (16 MB)
    __bf16* Kb    = (__bf16*)(ws + (size_t)167772160);   // 8x2048x128 (4 MB)
    __bf16* Vt    = (__bf16*)(ws + (size_t)171966464);   // 8x128x2048 (4 MB)
    __bf16* Ab    = (__bf16*)(ws + (size_t)176160768);   // 2048x4096 (16 MB)

    auto cvt = [&](const float* src, __bf16* dst, long long n) {
        long long blocks = (n / 4 + 255) / 256;
        cvt_f32_bf16<<<(unsigned)blocks, 256, 0, stream>>>(src, dst, n);
    };
    cvt(hidden, Xb, (long long)S_LEN * D_MODEL);
    cvt(wq, Wqkvb, (long long)D_MODEL * D_MODEL);
    cvt(wk, Wqkvb + (size_t)D_MODEL * D_MODEL, (long long)1024 * D_MODEL);
    cvt(wv, Wqkvb + (size_t)5120 * D_MODEL, (long long)1024 * D_MODEL);
    cvt(wo, Wob, (long long)D_MODEL * D_MODEL);

    dim3 g1(NQKV / 128, S_LEN / 128);
    gemm_bf16nt<<<g1, 256, 0, stream>>>(Xb, Wqkvb, QKV, S_LEN, NQKV, D_MODEL);

    long long tot = (long long)S_LEN * NQKV;
    rope_reorder<<<(unsigned)((tot + 255) / 256), 256, 0, stream>>>(QKV, Qb, Kb, Vt);

    dim3 ga(S_LEN / 16, NH);
    attn_fwd<<<ga, 32, 0, stream>>>(Qb, Kb, Vt, Ab);

    dim3 g2(D_MODEL / 128, S_LEN / 128);
    gemm_bf16nt<<<g2, 256, 0, stream>>>(Ab, Wob, (float*)d_out, S_LEN, D_MODEL, D_MODEL);
}
